// CustomMamba_11845519802613
// MI455X (gfx1250) — matrix operand order repaired and checked by reading, not compiled, not run on hardware
//
#include <hip/hip_runtime.h>
#include <hip/hip_bf16.h>
#include <cstdint>

#define D_MODEL  1024
#define D_STATE  16
#define D_CONV   4
#define D_INNER  2048
#define DT_RANK  64
#define IN_SIZE  512
#define OUT_SIZE 512
#define BATCH    2048
#define RNN_STRIDE (D_INNER * (D_CONV + D_STATE))   // 40960 floats per batch row

typedef __attribute__((ext_vector_type(16))) __bf16        v16bf;
typedef __attribute__((ext_vector_type(8)))  float         v8f;
typedef __attribute__((ext_vector_type(4)))  unsigned int  u32x4;
typedef __attribute__((ext_vector_type(8)))  int           i32x8;
typedef __attribute__((ext_vector_type(4)))  int           i32x4;

union FragBF { v16bf v; unsigned u[8]; };

// -------------------------------------------------------------------------
// TDM: DMA one BNx32 bf16 tile (rows of 32 elements, row stride K elements)
// from global memory into LDS at lds_off, rows packed back-to-back
// (tile row n -> LDS offset n*64 bytes).  D# per CDNA5 ISA 08 section 8.
// -------------------------------------------------------------------------
__device__ __forceinline__ void tdm_load_b_tile(const __bf16* gptr, unsigned lds_off,
                                                int rows, int K, int N)
{
  const uint64_t ga = (uint64_t)(uintptr_t)gptr;
  u32x4 g0;
  g0[0] = 1u;                                               // count=1, user mode
  g0[1] = lds_off;                                          // lds_addr [63:32]
  g0[2] = (unsigned)ga;                                     // global_addr lo
  g0[3] = ((unsigned)(ga >> 32) & 0x01FFFFFFu) | (2u << 30);// addr hi | type=2
  i32x8 g1;
  g1[0] = (int)(1u << 16);                                  // data_size=1 (2 bytes)
  g1[1] = (int)(((unsigned)K & 0xFFFFu) << 16);             // tensor_dim0 lo16
  g1[2] = (int)(((unsigned)K >> 16) | (((unsigned)N & 0xFFFFu) << 16)); // dim0 hi | dim1 lo
  g1[3] = (int)(((unsigned)N >> 16) | (32u << 16));         // dim1 hi | tile_dim0=32
  g1[4] = rows;                                             // tile_dim1
  g1[5] = (int)(unsigned)K;                                 // tensor_dim0_stride lo32
  g1[6] = 0;
  g1[7] = 0;
  const i32x4 z4 = {0, 0, 0, 0};                            // 2D: groups 2/3 unused
  const i32x8 z8 = {0, 0, 0, 0, 0, 0, 0, 0};                // extra group (6-arg form)
  __builtin_amdgcn_tensor_load_to_lds(g0, g1, z4, z4, z8, 0);
}

// -------------------------------------------------------------------------
// WMMA GEMM: C[M,N] = A[M,K](fp32) * W.T + (bias), W is N x K bf16 row-major.
// Block: 256 threads (8 waves, 4(M) x 2(N)).  Block tile (64*SM) x (32*SN),
// each wave computes an SMxSN grid of 16x16 tiles, K-step 32.
// B tiles arrive in LDS via the Tensor Data Mover (wave 0 issues, all wait).
// -------------------------------------------------------------------------
template<int SM, int SN, bool BIAS>
__global__ __launch_bounds__(256) void gemm_wmma_bf16(
    const float* __restrict__ A, const __bf16* __restrict__ W,
    const float* __restrict__ bias, float* __restrict__ C,
    int N, int K, int lda, int ldc)
{
  constexpr int BM  = 64 * SM;        // block rows
  constexpr int BN  = 32 * SN;        // block cols
  constexpr int KPT = BM / 8;         // A k-elements staged per thread
  constexpr int TPR = 32 / KPT;       // threads per A row

  __shared__ __bf16 As[BM][36];       // [m][k], padded
  __shared__ __bf16 Bs[BN][32];       // [n][k], packed (TDM fills)

  const int tid  = threadIdx.x;
  const int lane = tid & 31;
  const int wave = tid >> 5;
  const int wm   = wave >> 1;         // 0..3
  const int wn   = wave & 1;          // 0..1
  const int hi   = lane >> 4;
  const int l15  = lane & 15;

  const int n0 = blockIdx.x * BN;
  const int m0 = blockIdx.y * BM;

  const unsigned ldsb = (unsigned)(uintptr_t)&Bs[0][0];

  v8f acc[SM][SN] = {};

  for (int k0 = 0; k0 < K; k0 += 32) {
    // ---- B tile via Tensor Data Mover (one issue per workgroup) ----
    if (wave == 0) {
      tdm_load_b_tile(W + (size_t)n0 * K + k0, ldsb, BN, K, N);
      __builtin_amdgcn_s_wait_tensorcnt(0);
    }
    // ---- stage A tile BMx32: fp32 -> bf16 ----
    {
      const int m  = tid / TPR;
      const int kb = (tid % TPR) * KPT;
      const float* src = A + (size_t)(m0 + m) * lda + (k0 + kb);
#pragma unroll
      for (int c = 0; c < KPT / 4; ++c) {
        float4 f = *(const float4*)(src + 4 * c);
        As[m][kb + 4 * c + 0] = (__bf16)f.x;
        As[m][kb + 4 * c + 1] = (__bf16)f.y;
        As[m][kb + 4 * c + 2] = (__bf16)f.z;
        As[m][kb + 4 * c + 3] = (__bf16)f.w;
      }
    }
    __syncthreads();

    // ---- fragments per CDNA5 WMMA VGPR layouts ----
    FragBF a[SM], b[SN];
#pragma unroll
    for (int i = 0; i < 8; ++i) {
      const int ka  = (i < 4) ? (2 * i + 8 * hi) : (2 * i + 8 + 8 * hi);
      const int kb2 = (i < 4) ? (2 * i + 8 * hi) : (2 * i + 8 + 8 * hi);
#pragma unroll
      for (int si = 0; si < SM; ++si)
        a[si].u[i] = *(const unsigned*)&As[(wm * SM + si) * 16 + l15][ka];
#pragma unroll
      for (int sj = 0; sj < SN; ++sj)
        b[sj].u[i] = *(const unsigned*)&Bs[(wn * SN + sj) * 16 + l15][kb2];
    }
#pragma unroll
    for (int si = 0; si < SM; ++si)
#pragma unroll
      for (int sj = 0; sj < SN; ++sj)
        acc[si][sj] = __builtin_amdgcn_wmma_f32_16x16x32_bf16(
            false, a[si].v, false, b[sj].v, (short)0, acc[si][sj], false, false);
    __syncthreads();
  }

  // ---- store: VGPR r -> row r + 8*hi, col l15 ----
#pragma unroll
  for (int si = 0; si < SM; ++si) {
#pragma unroll
    for (int sj = 0; sj < SN; ++sj) {
      const int col = n0 + (wn * SN + sj) * 16 + l15;
      const float bv = BIAS ? bias[col] : 0.0f;
#pragma unroll
      for (int r = 0; r < 8; ++r) {
        const int row = m0 + (wm * SM + si) * 16 + r + 8 * hi;
        C[(size_t)row * ldc + col] = acc[si][sj][r] + bv;
      }
    }
  }
}

// -------------------------------------------------------------------------
// Weight pre-conversion: fp32 -> bf16 (and optional transpose to N x K).
// -------------------------------------------------------------------------
__global__ __launch_bounds__(256) void convert_bf16_kernel(
    const float* __restrict__ src, __bf16* __restrict__ dst, int n)
{
  const int i = blockIdx.x * blockDim.x + threadIdx.x;
  if (i < n) dst[i] = (__bf16)src[i];
}

__global__ __launch_bounds__(256) void convert_transpose_bf16_kernel(
    const float* __restrict__ src, __bf16* __restrict__ dst, int K, int N)
{ // src: K x N row-major -> dst: N x K row-major
  const int k = blockIdx.x * blockDim.x + threadIdx.x;
  const int n = blockIdx.y;
  if (k < K) dst[(size_t)n * K + k] = (__bf16)src[(size_t)k * N + n];
}

// -------------------------------------------------------------------------
// Conv state shift + depthwise conv + SiLU. One thread per (b, d).
// -------------------------------------------------------------------------
__global__ __launch_bounds__(256) void conv_silu_kernel(
    const float* __restrict__ rnn_in, const float* __restrict__ xz,
    const float* __restrict__ conv_w, const float* __restrict__ conv_b,
    float* __restrict__ rnn_out, float* __restrict__ xc)
{
  const int idx = blockIdx.x * blockDim.x + threadIdx.x;
  if (idx >= BATCH * D_INNER) return;
  const int b = idx >> 11;
  const int d = idx & (D_INNER - 1);

  const float* cs = rnn_in + (size_t)b * RNN_STRIDE + (size_t)d * (D_CONV + D_STATE);
  const float c1 = cs[1], c2 = cs[2], c3 = cs[3];
  const float xi = xz[(size_t)b * (2 * D_INNER) + d];

  const float* w = conv_w + (size_t)d * D_CONV;
  const float s = c1 * w[0] + c2 * w[1] + c3 * w[2] + xi * w[3] + conv_b[d];
  xc[idx] = s / (1.0f + expf(-s));                          // silu

  float4 st = make_float4(c1, c2, c3, xi);                  // shifted conv state
  *(float4*)(rnn_out + (size_t)b * RNN_STRIDE + (size_t)d * (D_CONV + D_STATE)) = st;
}

// -------------------------------------------------------------------------
// Selective-scan state update. One thread per (b, d).
// -------------------------------------------------------------------------
__global__ __launch_bounds__(256) void ssm_kernel(
    const float* __restrict__ rnn_in, const float* __restrict__ xz,
    const float* __restrict__ xdb, const float* __restrict__ dt_pre,
    const float* __restrict__ A_log, const float* __restrict__ Dp,
    float* __restrict__ rnn_out, float* xc_y)
{
  const int idx = blockIdx.x * blockDim.x + threadIdx.x;
  if (idx >= BATCH * D_INNER) return;
  const int b = idx >> 11;
  const int d = idx & (D_INNER - 1);

  const float xcv = xc_y[idx];
  const float dtp = dt_pre[idx];
  const float dt  = (dtp > 20.0f) ? dtp : log1pf(expf(dtp));   // softplus
  const float dtxc = dt * xcv;

  const float* xrow = xdb + (size_t)b * (DT_RANK + 2 * D_STATE);
  const float* al   = A_log + (size_t)d * D_STATE;
  const float* sin_ = rnn_in  + (size_t)b * RNN_STRIDE + (size_t)d * (D_CONV + D_STATE) + D_CONV;
  float*       sout = rnn_out + (size_t)b * RNN_STRIDE + (size_t)d * (D_CONV + D_STATE) + D_CONV;

  float acc = 0.0f;
#pragma unroll
  for (int n = 0; n < D_STATE; ++n) {
    const float Av   = -expf(al[n]);
    const float dA   = expf(dt * Av);
    const float snew = sin_[n] * dA + dtxc * xrow[DT_RANK + n];
    acc += snew * xrow[DT_RANK + D_STATE + n];
    sout[n] = snew;
  }

  const float zv = xz[(size_t)b * (2 * D_INNER) + D_INNER + d];
  const float g  = zv / (1.0f + expf(-zv));                    // silu(z)
  xc_y[idx] = (acc + Dp[d] * xcv) * g;                         // y (in place)
}

// -------------------------------------------------------------------------
extern "C" void kernel_launch(void* const* d_in, const int* in_sizes, int n_in,
                              void* d_out, int out_size, void* d_ws, size_t ws_size,
                              hipStream_t stream)
{
  (void)in_sizes; (void)n_in; (void)out_size; (void)ws_size;

  const float* x          = (const float*)d_in[0];
  const float* rnn_in     = (const float*)d_in[1];
  const float* w_inp      = (const float*)d_in[2];
  const float* b_inp      = (const float*)d_in[3];
  const float* w_outp     = (const float*)d_in[4];
  const float* b_outp     = (const float*)d_in[5];
  const float* in_proj_w  = (const float*)d_in[6];
  const float* conv_w     = (const float*)d_in[7];
  const float* conv_b     = (const float*)d_in[8];
  const float* x_proj_w   = (const float*)d_in[9];
  const float* dt_proj_w  = (const float*)d_in[10];
  const float* dt_bias    = (const float*)d_in[11];
  const float* A_log      = (const float*)d_in[12];
  const float* Dp         = (const float*)d_in[13];
  const float* out_proj_w = (const float*)d_in[14];

  // ---- workspace: fp32 activations, then bf16 weights ----
  float* ws     = (float*)d_ws;
  float* h      = ws;                                          // 2048*1024
  float* xz     = h   + (size_t)BATCH * D_MODEL;               // 2048*4096
  float* xc     = xz  + (size_t)BATCH * 2 * D_INNER;           // 2048*2048 (also y)
  float* xdb    = xc  + (size_t)BATCH * D_INNER;               // 2048*96
  float* dt_pre = xdb + (size_t)BATCH * (DT_RANK + 2 * D_STATE); // 2048*2048
  float* core   = h;

  __bf16* w_inp_b   = (__bf16*)(dt_pre + (size_t)BATCH * D_INNER);
  __bf16* w_outp_b  = w_inp_b   + (size_t)D_MODEL * IN_SIZE;   // 1024x512
  __bf16* in_proj_b = w_outp_b  + (size_t)OUT_SIZE * D_MODEL;  // 512x1024
  __bf16* x_proj_b  = in_proj_b + (size_t)2 * D_INNER * D_MODEL;
  __bf16* dt_proj_b = x_proj_b  + (size_t)(DT_RANK + 2 * D_STATE) * D_INNER;
  __bf16* out_proj_b= dt_proj_b + (size_t)D_INNER * DT_RANK;

  float* out     = (float*)d_out;                              // 2048*512
  float* rnn_out = out + (size_t)BATCH * OUT_SIZE;             // 2048*40960

  const dim3 blk(256);
  const dim3 ew((BATCH * D_INNER) / 256);

  // ---- weight pre-conversion to bf16, all stored N x K row-major ----
  convert_transpose_bf16_kernel<<<dim3(IN_SIZE / 256, D_MODEL), blk, 0, stream>>>(
      w_inp, w_inp_b, IN_SIZE, D_MODEL);
  convert_transpose_bf16_kernel<<<dim3(D_MODEL / 256, OUT_SIZE), blk, 0, stream>>>(
      w_outp, w_outp_b, D_MODEL, OUT_SIZE);
  convert_bf16_kernel<<<dim3((2 * D_INNER * D_MODEL) / 256), blk, 0, stream>>>(
      in_proj_w, in_proj_b, 2 * D_INNER * D_MODEL);
  convert_bf16_kernel<<<dim3(((DT_RANK + 2 * D_STATE) * D_INNER) / 256), blk, 0, stream>>>(
      x_proj_w, x_proj_b, (DT_RANK + 2 * D_STATE) * D_INNER);
  convert_bf16_kernel<<<dim3((D_INNER * DT_RANK) / 256), blk, 0, stream>>>(
      dt_proj_w, dt_proj_b, D_INNER * DT_RANK);
  convert_bf16_kernel<<<dim3((D_MODEL * D_INNER) / 256), blk, 0, stream>>>(
      out_proj_w, out_proj_b, D_MODEL * D_INNER);

  // 1) h = x @ w_inp + b_inp
  gemm_wmma_bf16<2, 2, true><<<dim3(D_MODEL / 64, BATCH / 128), blk, 0, stream>>>(
      x, w_inp_b, b_inp, h, D_MODEL, IN_SIZE, IN_SIZE, D_MODEL);
  // 2) xz = h @ in_proj_w.T
  gemm_wmma_bf16<2, 2, false><<<dim3((2 * D_INNER) / 64, BATCH / 128), blk, 0, stream>>>(
      h, in_proj_b, nullptr, xz, 2 * D_INNER, D_MODEL, D_MODEL, 2 * D_INNER);
  // 3) conv shift + silu -> xc ; new conv_state -> rnn_out
  conv_silu_kernel<<<ew, blk, 0, stream>>>(rnn_in, xz, conv_w, conv_b, rnn_out, xc);
  // 4) xdb = xc @ x_proj_w.T   (N=96 -> small-tile variant)
  gemm_wmma_bf16<1, 1, false><<<dim3((DT_RANK + 2 * D_STATE) / 32, BATCH / 64), blk, 0, stream>>>(
      xc, x_proj_b, nullptr, xdb, DT_RANK + 2 * D_STATE, D_INNER, D_INNER,
      DT_RANK + 2 * D_STATE);
  // 5) dt_pre = xdb[:, :64] @ dt_proj_w.T + dt_proj_b
  gemm_wmma_bf16<2, 2, true><<<dim3(D_INNER / 64, BATCH / 128), blk, 0, stream>>>(
      xdb, dt_proj_b, dt_bias, dt_pre, D_INNER, DT_RANK, DT_RANK + 2 * D_STATE, D_INNER);
  // 6) SSM update: new ssm_state -> rnn_out ; y -> xc (in place)
  ssm_kernel<<<ew, blk, 0, stream>>>(rnn_in, xz, xdb, dt_pre, A_log, Dp, rnn_out, xc);
  // 7) core = y @ out_proj_w.T
  gemm_wmma_bf16<2, 2, false><<<dim3(D_MODEL / 64, BATCH / 128), blk, 0, stream>>>(
      xc, out_proj_b, nullptr, core, D_MODEL, D_INNER, D_INNER, D_MODEL);
  // 8) out = core @ w_outp + b_outp
  gemm_wmma_bf16<2, 2, true><<<dim3(OUT_SIZE / 64, BATCH / 128), blk, 0, stream>>>(
      core, w_outp_b, b_outp, out, OUT_SIZE, D_MODEL, D_MODEL, OUT_SIZE);
}
